// GATClassifier_50044958933477
// MI455X (gfx1250) — matrix operand order repaired and checked by reading, not compiled, not run on hardware
//
#include <hip/hip_runtime.h>
#include <hip/hip_bf16.h>
#include <math.h>

#define NN   50000
#define EE   400000
#define ETOT (EE + NN)      // edges + self loops
#define GG   16
#define DD   128
#define HH   4
#define C1V  128
#define C2V  64
#define HC1  (HH * C1V)     // 512
#define HC2  (HH * C2V)     // 256

typedef __bf16 bf16;
typedef __attribute__((ext_vector_type(16))) __bf16 v16bf;
typedef __attribute__((ext_vector_type(8)))  __bf16 v8bf;
typedef __attribute__((ext_vector_type(8)))  float  v8f;

// ---------- helpers ----------
__device__ __forceinline__ unsigned fenc(float f) {
    unsigned u = __float_as_uint(f);
    return (u & 0x80000000u) ? ~u : (u | 0x80000000u);
}
__device__ __forceinline__ float fdec(unsigned e) {
    unsigned u = (e & 0x80000000u) ? (e & 0x7FFFFFFFu) : ~e;
    return __uint_as_float(u);
}
__device__ __forceinline__ void edge_sd(const int* __restrict__ ei, long e, int& s, int& d) {
    if (e < EE) { s = ei[e]; d = ei[EE + e]; }
    else        { s = d = (int)(e - EE); }
}
__device__ __forceinline__ float eluf(float v) { return v > 0.f ? v : expm1f(v); }

// ---------- pack fp32 weight [K,NC] into WMMA B-fragment order, bf16 ----------
// Bp[((nt*(K/32)+ks)*32 + lane)*16 + j] = W[(ks*32 + (lane>>4)*8 + (j&7) + (j>>3)*16)*NC + nt*16 + (lane&15)]
template <int K, int NC>
__global__ void k_packB(const float* __restrict__ W, bf16* __restrict__ Bp) {
    int t = blockIdx.x * blockDim.x + threadIdx.x;
    if (t >= K * NC) return;
    int j    = t & 15;
    int lane = (t >> 4) & 31;
    int rest = t >> 9;               // nt*(K/32) + ks
    int ks   = rest % (K / 32);
    int nt   = rest / (K / 32);
    int col  = nt * 16 + (lane & 15);
    int row  = ks * 32 + (lane >> 4) * 8 + (j & 7) + (j >> 3) * 16;
    Bp[t] = (bf16)W[(size_t)row * NC + col];
}

// ---------- ce[h] = sum_c lin_e[h*C+c] * att_e[h*C+c] ----------
__global__ void k_ce(const float* __restrict__ We, const float* __restrict__ Ae,
                     float* __restrict__ ce, int C) {
    int h = threadIdx.x;
    if (h < HH) {
        float s = 0.f;
        for (int c = 0; c < C; ++c) s += We[h * C + c] * Ae[h * C + c];
        ce[h] = s;
    }
}

// ---------- self-loop fill_value='mean': deg + attr sum per dst ----------
__global__ void k_selfloop(const int* __restrict__ ei, const float* __restrict__ eattr,
                           float* __restrict__ deg, float* __restrict__ asum) {
    int t = blockIdx.x * blockDim.x + threadIdx.x;
    if (t < EE) {
        int d = ei[EE + t];
        atomicAdd(&deg[d], 1.0f);
        atomicAdd(&asum[d], eattr[t]);
    }
}

// ---------- WMMA GEMM: C[M,NC] = A[M,K] * Bp(pre-packed), fp32 acc ----------
// 256 threads = 8 waves; each wave owns TPW 16-wide column tiles; block owns 16 rows.
template <typename AT, int K, int NC, int TPW>
__global__ void __launch_bounds__(256)
k_wmma_gemm(const AT* __restrict__ A, const bf16* __restrict__ Bp, float* __restrict__ C) {
    const int lane = threadIdx.x & 31;
    const int wv   = threadIdx.x >> 5;
    const int row0 = blockIdx.x * 16;
    const int rA   = row0 + (lane & 15);
    const int hiA  = (lane >> 4) * 8;    // A K-offset group per ISA layout

    v8f acc[TPW];
    #pragma unroll
    for (int t = 0; t < TPW; ++t)
        #pragma unroll
        for (int i = 0; i < 8; ++i) acc[t][i] = 0.f;

    for (int kk = 0; kk < K; kk += 32) {
        // A fragment: lanes 0-15 -> K kk+[0..7], kk+[16..23]; lanes 16-31 -> +8
        v16bf a;
        const AT* ap = A + (size_t)rA * K + kk + hiA;
        if constexpr (sizeof(AT) == 4) {
            const float4 p0 = *reinterpret_cast<const float4*>(ap + 0);
            const float4 p1 = *reinterpret_cast<const float4*>(ap + 4);
            const float4 p2 = *reinterpret_cast<const float4*>(ap + 16);
            const float4 p3 = *reinterpret_cast<const float4*>(ap + 20);
            a[0] = (bf16)p0.x;  a[1] = (bf16)p0.y;  a[2]  = (bf16)p0.z;  a[3]  = (bf16)p0.w;
            a[4] = (bf16)p1.x;  a[5] = (bf16)p1.y;  a[6]  = (bf16)p1.z;  a[7]  = (bf16)p1.w;
            a[8] = (bf16)p2.x;  a[9] = (bf16)p2.y;  a[10] = (bf16)p2.z;  a[11] = (bf16)p2.w;
            a[12] = (bf16)p3.x; a[13] = (bf16)p3.y; a[14] = (bf16)p3.z;  a[15] = (bf16)p3.w;
        } else {
            const v8bf lo = *reinterpret_cast<const v8bf*>(ap);
            const v8bf hi = *reinterpret_cast<const v8bf*>(ap + 16);
            #pragma unroll
            for (int j = 0; j < 8; ++j) { a[j] = lo[j]; a[8 + j] = hi[j]; }
        }

        #pragma unroll
        for (int t = 0; t < TPW; ++t) {
            const int ntg = wv * TPW + t;                      // global 16-col tile
            const size_t frag = (size_t)(ntg * (K / 32) + (kk >> 5));
            const v16bf b = *reinterpret_cast<const v16bf*>(Bp + (frag * 32 + lane) * 16);
            acc[t] = __builtin_amdgcn_wmma_f32_16x16x32_bf16(
                false, a, false, b, (short)0, acc[t], false, false);
        }
    }

    // C layout: VGPR r -> row r (lanes 0-15) / r+8 (lanes 16-31), N = lane&15
    const int rOut = row0 + (lane >> 4) * 8;
    #pragma unroll
    for (int t = 0; t < TPW; ++t) {
        const int col = (wv * TPW + t) * 16 + (lane & 15);
        #pragma unroll
        for (int r = 0; r < 8; ++r)
            C[(size_t)(rOut + r) * NC + col] = acc[t][r];
    }
}

// ---------- per-(node,head) attention scores; one wave per pair ----------
template <int C>
__global__ void k_scores(const float* __restrict__ h, const float* __restrict__ a_src,
                         const float* __restrict__ a_dst, float* __restrict__ s_src,
                         float* __restrict__ s_dst) {
    const int lane = threadIdx.x & 31;
    const int wv   = threadIdx.x >> 5;
    long pair = (long)blockIdx.x * 8 + wv;
    if (pair >= (long)NN * HH) return;
    int n = (int)(pair >> 2), hh = (int)(pair & 3);
    const float* hp  = h + (size_t)n * (HH * C) + hh * C;
    const float* asp = a_src + hh * C;
    const float* adp = a_dst + hh * C;
    float ss = 0.f, sd = 0.f;
    for (int i = lane; i < C; i += 32) {
        float v = hp[i];
        ss += v * asp[i];
        sd += v * adp[i];
    }
    #pragma unroll
    for (int off = 16; off; off >>= 1) {
        ss += __shfl_xor(ss, off, 32);
        sd += __shfl_xor(sd, off, 32);
    }
    if (lane == 0) { s_src[n * HH + hh] = ss; s_dst[n * HH + hh] = sd; }
}

// ---------- pass A: alpha = leaky_relu(...) ; atomic segment max (encoded) ----------
__global__ void k_alpha(const int* __restrict__ ei, const float* __restrict__ eattr,
                        const float* __restrict__ deg, const float* __restrict__ asum,
                        const float* __restrict__ s_src, const float* __restrict__ s_dst,
                        const float* __restrict__ ce, float* __restrict__ abuf,
                        unsigned* __restrict__ amax) {
    long t = (long)blockIdx.x * blockDim.x + threadIdx.x;
    if (t >= (long)ETOT * HH) return;
    long e = t >> 2; int hh = (int)(t & 3);
    int s, d; edge_sd(ei, e, s, d);
    float ea = (e < EE) ? eattr[e] : asum[d] / fmaxf(deg[d], 1.0f);
    float a = s_src[s * HH + hh] + s_dst[d * HH + hh] + ea * ce[hh];
    a = (a > 0.f) ? a : 0.2f * a;          // leaky_relu 0.2
    abuf[t] = a;
    atomicMax(&amax[d * HH + hh], fenc(a));
}

// ---------- pass B: ex = exp(alpha - amax[dst]); denom += ex ----------
__global__ void k_exp(const int* __restrict__ ei, const unsigned* __restrict__ amax,
                      float* __restrict__ abuf, float* __restrict__ den) {
    long t = (long)blockIdx.x * blockDim.x + threadIdx.x;
    if (t >= (long)ETOT * HH) return;
    long e = t >> 2; int hh = (int)(t & 3);
    int s, d; edge_sd(ei, e, s, d);
    float ex = expf(abuf[t] - fdec(amax[d * HH + hh]));
    abuf[t] = ex;
    atomicAdd(&den[d * HH + hh], ex);
}

// ---------- pass C: out[dst] += (ex/denom) * h[src]  (HC/4 threads per edge) ----------
template <int HC, int C>
__global__ void k_agg(const int* __restrict__ ei, const float* __restrict__ abuf,
                      const float* __restrict__ den, const float* __restrict__ h,
                      float* __restrict__ out) {
    long t = (long)blockIdx.x * blockDim.x + threadIdx.x;
    const int TPE = HC / 4;
    long e = t / TPE;
    if (e >= ETOT) return;
    int c4 = (int)(t - e * TPE) * 4;
    int hh = c4 / C;
    int s, d; edge_sd(ei, e, s, d);
    float w = abuf[e * HH + hh] / (den[d * HH + hh] + 1e-16f);
    const float4 hv = *reinterpret_cast<const float4*>(h + (size_t)s * HC + c4);
    float* op = out + (size_t)d * HC + c4;
    atomicAdd(op + 0, w * hv.x);
    atomicAdd(op + 1, w * hv.y);
    atomicAdd(op + 2, w * hv.z);
    atomicAdd(op + 3, w * hv.w);
}

// ---------- conv1 epilogue: +b1, BN, ELU -> bf16 for GEMM2 ----------
__global__ void k_post1(const float* __restrict__ out1, const float* __restrict__ b1,
                        const float* __restrict__ g, const float* __restrict__ bb,
                        const float* __restrict__ m, const float* __restrict__ vv,
                        bf16* __restrict__ h1b) {
    long t = (long)blockIdx.x * blockDim.x + threadIdx.x;
    if (t >= (long)NN * HC1) return;
    int c = (int)(t & (HC1 - 1));
    float v = out1[t] + b1[c];
    v = (v - m[c]) * (g[c] * rsqrtf(vv[c] + 1e-5f)) + bb[c];
    h1b[t] = (bf16)eluf(v);
}

// ---------- conv2 epilogue: head-mean, +b2, BN, ELU, pooled atomicAdd ----------
__global__ void k_post2(const float* __restrict__ out2, const float* __restrict__ b2,
                        const float* __restrict__ g, const float* __restrict__ bb,
                        const float* __restrict__ m, const float* __restrict__ vv,
                        const int* __restrict__ batch, float* __restrict__ pool) {
    long t = (long)blockIdx.x * blockDim.x + threadIdx.x;
    if (t >= (long)NN * C2V) return;
    int n = (int)(t >> 6), c = (int)(t & 63);
    const float* op = out2 + (size_t)n * HC2;
    float v = 0.25f * (op[c] + op[C2V + c] + op[2 * C2V + c] + op[3 * C2V + c]);
    v += b2[c];
    v = (v - m[c]) * (g[c] * rsqrtf(vv[c] + 1e-5f)) + bb[c];
    atomicAdd(&pool[batch[n] * C2V + c], eluf(v));
}

__global__ void k_cnt(const int* __restrict__ batch, float* __restrict__ cnt) {
    int t = blockIdx.x * blockDim.x + threadIdx.x;
    if (t < NN) atomicAdd(&cnt[batch[t]], 1.0f);
}

// ---------- FC head: mean-pool normalize, fc1+ELU, fc2 ----------
__global__ void __launch_bounds__(512)
k_head(const float* __restrict__ pool, const float* __restrict__ cnt,
       const float* __restrict__ fc1W, const float* __restrict__ fc1b,
       const float* __restrict__ fc2W, const float* __restrict__ fc2b,
       float* __restrict__ out) {
    __shared__ float z[GG][32];
    int g = threadIdx.x >> 5, j = threadIdx.x & 31;
    float inv = 1.0f / fmaxf(cnt[g], 1.0f);
    float acc = fc1b[j];
    for (int c = 0; c < C2V; ++c) acc += (pool[g * C2V + c] * inv) * fc1W[c * 32 + j];
    z[g][j] = eluf(acc);
    __syncthreads();
    if (j == 0) {
        float o = fc2b[0];
        for (int k = 0; k < 32; ++k) o += z[g][k] * fc2W[k];
        out[g] = o;
    }
}

// =====================================================================
extern "C" void kernel_launch(void* const* d_in, const int* in_sizes, int n_in,
                              void* d_out, int out_size, void* d_ws, size_t ws_size,
                              hipStream_t stream) {
    const float* x      = (const float*)d_in[0];
    const int*   ei     = (const int*)d_in[1];
    const int*   batch  = (const int*)d_in[2];
    const float* eattr  = (const float*)d_in[3];
    const float* W1     = (const float*)d_in[4];
    const float* asrc1  = (const float*)d_in[5];
    const float* adst1  = (const float*)d_in[6];
    const float* line1  = (const float*)d_in[7];
    const float* atte1  = (const float*)d_in[8];
    const float* b1     = (const float*)d_in[9];
    const float* bn1g   = (const float*)d_in[10];
    const float* bn1b   = (const float*)d_in[11];
    const float* bn1m   = (const float*)d_in[12];
    const float* bn1v   = (const float*)d_in[13];
    const float* W2     = (const float*)d_in[14];
    const float* asrc2  = (const float*)d_in[15];
    const float* adst2  = (const float*)d_in[16];
    const float* line2  = (const float*)d_in[17];
    const float* atte2  = (const float*)d_in[18];
    const float* b2     = (const float*)d_in[19];
    const float* bn2g   = (const float*)d_in[20];
    const float* bn2b   = (const float*)d_in[21];
    const float* bn2m   = (const float*)d_in[22];
    const float* bn2v   = (const float*)d_in[23];
    const float* fc1W   = (const float*)d_in[24];
    const float* fc1b   = (const float*)d_in[25];
    const float* fc2W   = (const float*)d_in[26];
    const float* fc2b   = (const float*)d_in[27];
    float* outp = (float*)d_out;

    // ---- workspace carving ----
    char* ws = (char*)d_ws;
    size_t off = 0;
    auto take = [&](size_t bytes) -> void* {
        void* p = ws + off;
        off += (bytes + 255) & ~(size_t)255;
        return p;
    };
    // zeroed-at-start block (contiguous, one memset)
    float*    deg   = (float*)take((size_t)NN * 4);
    float*    asum  = (float*)take((size_t)NN * 4);
    unsigned* amax1 = (unsigned*)take((size_t)NN * HH * 4);
    float*    den1  = (float*)take((size_t)NN * HH * 4);
    unsigned* amax2 = (unsigned*)take((size_t)NN * HH * 4);
    float*    den2  = (float*)take((size_t)NN * HH * 4);
    float*    pool  = (float*)take((size_t)GG * C2V * 4);
    float*    cnt   = (float*)take((size_t)GG * 4);
    size_t zeroEnd = off;
    // big buffers (reused across layers)
    float* out1 = (float*)take((size_t)NN * HC1 * 4);   // conv1 agg out; reused as out2
    float* h1   = (float*)take((size_t)NN * HC1 * 4);   // gemm1 out; reused as h2
    bf16*  h1b  = (bf16*) take((size_t)NN * HC1 * 2);   // post-BN bf16 activations
    float* abuf = (float*)take((size_t)ETOT * HH * 4);  // alpha / exp per edge-head
    bf16*  W1p  = (bf16*) take((size_t)DD * HC1 * 2);   // fragment-packed bf16 weights
    bf16*  W2p  = (bf16*) take((size_t)HC1 * HC2 * 2);
    float* ce1  = (float*)take(HH * 4);
    float* ce2  = (float*)take(HH * 4);
    float* as1  = (float*)take((size_t)NN * HH * 4);
    float* ad1  = (float*)take((size_t)NN * HH * 4);
    float* as2  = (float*)take((size_t)NN * HH * 4);
    float* ad2  = (float*)take((size_t)NN * HH * 4);
    float* out2 = out1;
    float* h2   = h1;
    (void)ws_size; (void)n_in; (void)in_sizes; (void)out_size;

    // ---- zero accumulators ----
    hipMemsetAsync(ws, 0, zeroEnd, stream);
    hipMemsetAsync(out1, 0, (size_t)NN * HC1 * 4, stream);

    // ---- prep: weight fragment-packing, edge-attr score constants, self-loops ----
    k_packB<DD, HC1><<<(DD * HC1 + 255) / 256, 256, 0, stream>>>(W1, W1p);
    k_packB<HC1, HC2><<<(HC1 * HC2 + 255) / 256, 256, 0, stream>>>(W2, W2p);
    k_ce<<<1, 32, 0, stream>>>(line1, atte1, ce1, C1V);
    k_ce<<<1, 32, 0, stream>>>(line2, atte2, ce2, C2V);
    k_selfloop<<<(EE + 255) / 256, 256, 0, stream>>>(ei, eattr, deg, asum);

    // ---- conv1 ----
    k_wmma_gemm<float, DD, HC1, 4><<<NN / 16, 256, 0, stream>>>(x, W1p, h1);
    k_scores<C1V><<<(NN * HH + 7) / 8, 256, 0, stream>>>(h1, asrc1, adst1, as1, ad1);
    {
        long tot = (long)ETOT * HH;
        int blks = (int)((tot + 255) / 256);
        k_alpha<<<blks, 256, 0, stream>>>(ei, eattr, deg, asum, as1, ad1, ce1, abuf, amax1);
        k_exp<<<blks, 256, 0, stream>>>(ei, amax1, abuf, den1);
    }
    {
        long tot = (long)ETOT * (HC1 / 4);
        k_agg<HC1, C1V><<<(int)((tot + 255) / 256), 256, 0, stream>>>(ei, abuf, den1, h1, out1);
    }
    k_post1<<<(int)(((long)NN * HC1 + 255) / 256), 256, 0, stream>>>(
        out1, b1, bn1g, bn1b, bn1m, bn1v, h1b);

    // ---- conv2 (re-zero aliased out region first) ----
    hipMemsetAsync(out2, 0, (size_t)NN * HC2 * 4, stream);
    k_wmma_gemm<bf16, HC1, HC2, 2><<<NN / 16, 256, 0, stream>>>(h1b, W2p, h2);
    k_scores<C2V><<<(NN * HH + 7) / 8, 256, 0, stream>>>(h2, asrc2, adst2, as2, ad2);
    {
        long tot = (long)ETOT * HH;
        int blks = (int)((tot + 255) / 256);
        k_alpha<<<blks, 256, 0, stream>>>(ei, eattr, deg, asum, as2, ad2, ce2, abuf, amax2);
        k_exp<<<blks, 256, 0, stream>>>(ei, amax2, abuf, den2);
    }
    {
        long tot = (long)ETOT * (HC2 / 4);
        k_agg<HC2, C2V><<<(int)((tot + 255) / 256), 256, 0, stream>>>(ei, abuf, den2, h2, out2);
    }

    // ---- pooling + head ----
    k_post2<<<(int)(((long)NN * C2V + 255) / 256), 256, 0, stream>>>(
        out2, b2, bn2g, bn2b, bn2m, bn2v, batch, pool);
    k_cnt<<<(NN + 255) / 256, 256, 0, stream>>>(batch, cnt);
    k_head<<<1, 512, 0, stream>>>(pool, cnt, fc1W, fc1b, fc2W, fc2b, outp);
}
